// BusDecoder_32701880992367
// MI455X (gfx1250) — compile-verified, matrix operand order checked by reading, not run on hardware
//
#include <hip/hip_runtime.h>

// ---------------------------------------------------------------------------
// BusDecoder on MI455X (gfx1250, wave32, WMMA)
//
// out[b, n*2+o] = sum_d H[b,n,d] * W[t[n], d, o] + bias[t[n], o],  t[n]=bus_type[0,n]
//
// HBM-bound: 512 MB of H vs 0.54 GFLOP -> floor ~22us @ 23.3 TB/s.
// Strategy: uniform GEMM [B*N, 1024] x [1024, 16] (cols 0..5 = the 3 decoders'
// 6 columns, cols 6..15 = 0) with v_wmma_f32_16x16x32_bf16, then per-row
// column-select + bias at store time. B matrix (32KB bf16) lives in LDS,
// pre-swizzled to the WMMA B-operand per-lane layout.
// ---------------------------------------------------------------------------

typedef __attribute__((ext_vector_type(16))) __bf16 v16bf;
typedef __attribute__((ext_vector_type(8)))  float  v8f;
typedef __attribute__((ext_vector_type(4)))  float  v4f;

#define DDIM   1024
#define NPOS   4096
#define NBATCH 32
#define NTYPES 3
#define KSTEP  32
#define NSLICE (DDIM / KSTEP)      // 32 K-slices
#define WAVES  8                   // waves per block (256 threads)

__global__ __launch_bounds__(256) void bus_decoder_wmma(
    const float* __restrict__ H,        // [32, 4096, 1024] f32
    const int*   __restrict__ bus_type, // [32, 4096] (row 0 used)
    const float* __restrict__ W,        // [3, 1024, 2] f32
    const float* __restrict__ bias,     // [3, 2] f32
    float*       __restrict__ out)      // [32, 8192] f32
{
    // B operand in LDS, pre-swizzled: for slice s, lane l holds 16 bf16
    // (its v16bf WMMA B operand) at sB[(s*32 + l)*16 .. +15].
    // Element e of lane l maps to (col = l&15, K = s*32 + kin(e, l>>4)).
    __shared__ __bf16 sB[NSLICE * 32 * 16];   // 32 KB

    const int tid = threadIdx.x;

    // ---- build swizzled B once per block: 16384 bf16 / 256 threads ----
    for (int idx = tid; idx < NSLICE * 32 * 16; idx += 256) {
        const int e     = idx & 15;
        const int l     = (idx >> 4) & 31;
        const int s     = idx >> 9;
        const int col   = l & 15;
        const int khalf = l >> 4;
        // 16-bit A/B VGPR layout: elems 0..7 -> K = khalf*8 + e,
        //                         elems 8..15 -> K = 16 + khalf*8 + (e-8)
        const int kin = (e < 8) ? (khalf * 8 + e) : (16 + khalf * 8 + (e - 8));
        const int K   = s * KSTEP + kin;
        float v = 0.0f;
        if (col < 2 * NTYPES) {                       // cols 0..5 = (t, o)
            v = W[(col >> 1) * (DDIM * 2) + K * 2 + (col & 1)];
        }
        sB[idx] = (__bf16)v;
    }
    __syncthreads();

    const int lane  = tid & 31;
    const int wave  = tid >> 5;
    const int khalf = lane >> 4;          // which half-wave (K grouping)
    const int mrow  = lane & 15;          // A-matrix row within tile

    const int tile = blockIdx.x * WAVES + wave;   // 8192 tiles of 16 rows
    const int r0   = tile * 16;                   // flattened (b,n) row base
    const int b    = r0 / NPOS;                   // whole tile shares b
    const int n0   = r0 % NPOS;

    const float* rowPtr = H + (size_t)(r0 + mrow) * DDIM;

    v8f acc = {};

    #pragma unroll 4
    for (int s = 0; s < NSLICE; ++s) {
        const int ks = s * KSTEP;

        // A operand: this lane needs K = ks + khalf*8 + [0..7]  (elems 0..7)
        //                     and K = ks + 16 + khalf*8 + [0..7] (elems 8..15)
        const float* p0 = rowPtr + ks + khalf * 8;
        const float* p1 = p0 + 16;
        v4f f0 = *(const v4f*)(p0);
        v4f f1 = *(const v4f*)(p0 + 4);
        v4f f2 = *(const v4f*)(p1);
        v4f f3 = *(const v4f*)(p1 + 4);

        // stream-ahead prefetch of H (global_prefetch_b8)
        if (s + 2 < NSLICE)
            __builtin_prefetch((const void*)(rowPtr + ks + 2 * KSTEP), 0, 1);

        float af[16];
        *(v4f*)(af + 0)  = f0;
        *(v4f*)(af + 4)  = f1;
        *(v4f*)(af + 8)  = f2;
        *(v4f*)(af + 12) = f3;

        v16bf a;
        #pragma unroll
        for (int e = 0; e < 16; ++e) a[e] = (__bf16)af[e];

        // B operand: 32B LDS read (2x ds_load_b128)
        const v16bf bm = *(const v16bf*)(&sB[(s * 32 + lane) * 16]);

        // D = A(16x32 bf16) * B(32x16 bf16) + C(16x16 f32)
        acc = __builtin_amdgcn_wmma_f32_16x16x32_bf16(
                  false, a, false, bm, (short)0, acc, false, false);
    }

    // ---- column-select + bias + store ----
    // acc VGPR v, lane l: row M = v + 8*(l>>4), column Ncol = l&15.
    // Row M has type t = bus_type[n0+M]; useful cols are 2t and 2t+1.
    const int colsel = lane & 15;
    #pragma unroll
    for (int v = 0; v < 8; ++v) {
        const int M = v + 8 * khalf;
        const int n = n0 + M;
        const int t = bus_type[n];            // row 0 of bus_type
        const int cbase = 2 * t;
        if (colsel == cbase || colsel == cbase + 1) {
            const int o = colsel - cbase;
            out[(size_t)b * (NPOS * 2) + (size_t)n * 2 + o] =
                acc[v] + bias[t * 2 + o];
        }
    }
}

extern "C" void kernel_launch(void* const* d_in, const int* in_sizes, int n_in,
                              void* d_out, int out_size, void* d_ws, size_t ws_size,
                              hipStream_t stream) {
    const float* H        = (const float*)d_in[0];
    const int*   bus_type = (const int*)d_in[1];
    const float* W        = (const float*)d_in[2];
    const float* bias     = (const float*)d_in[3];
    float*       out      = (float*)d_out;

    // 8192 row-tiles of 16, one tile per wave, 8 waves per block
    const int tiles  = (NBATCH * NPOS) / 16;   // 8192
    const int blocks = tiles / WAVES;          // 1024
    bus_decoder_wmma<<<blocks, 256, 0, stream>>>(H, bus_type, W, bias, out);
}
